// MultiHeadedAttentionRelative_55817394979634
// MI455X (gfx1250) — matrix operand order repaired and checked by reading, not compiled, not run on hardware
//
// MI455X (gfx1250) multi-head attention with relative position encodings.
// bf16 WMMA (v_wmma_f32_16x16x32_bf16) for all six contractions, f32 accum.
// rel_k / rel_v streamed exactly once via per-q (64 x K x 1024) GEMMs.
// Projection GEMMs use gfx1250 async global->LDS copies (ASYNCcnt) with
// double buffering; K-loops carry global_prefetch_b8 hints.
#include <hip/hip_runtime.h>

typedef __bf16 bf16;
typedef __attribute__((ext_vector_type(16))) __bf16 v16bf;
typedef __attribute__((ext_vector_type(8)))  __bf16 v8bf;
typedef __attribute__((ext_vector_type(8)))  float  v8f;

static constexpr int Bn = 4, Ln = 1024, Dn = 1024, Hn = 16, DHn = 64, BHn = 64;

// ---------------- gfx1250 async global -> LDS copy (ASYNCcnt) ---------------
__device__ __forceinline__ void async_b128(unsigned lds_off, const bf16* g) {
  asm volatile("global_load_async_to_lds_b128 %0, %1, off"
               :: "v"(lds_off), "v"((unsigned long long)(size_t)g)
               : "memory");
}
__device__ __forceinline__ void wait_async0() {
  asm volatile("s_wait_asynccnt 0" ::: "memory");
}

// ---------------- WMMA fragment loaders (wave32, 16x16x32 bf16) -------------
// A fragment 16x32 (MxK), row-major A. Lane l owns row m=(l&15);
// element e holds k = (e>>3)*16 + ((l>>4)<<3) + (e&7).
__device__ __forceinline__ v16bf frag_a(const bf16* __restrict__ p0, long lda) {
  int lane = threadIdx.x & 31;
  const bf16* p = p0 + (long)(lane & 15) * lda + ((lane >> 4) << 3);
  v8bf lo = *(const v8bf*)(p);        // k +0..7
  v8bf hi = *(const v8bf*)(p + 16);   // k +16..23
  return __builtin_shufflevector(lo, hi, 0,1,2,3,4,5,6,7,8,9,10,11,12,13,14,15);
}
// B fragment 32x16 (KxN) where B[k][n] = Bt[n][k] (Bt row-major NxK, ld=ldb).
// Lane l owns col n=(l&15); element e holds k = (e>>3)*16 + ((l>>4)<<3) + (e&7).
__device__ __forceinline__ v16bf frag_b(const bf16* __restrict__ p0, long ldb) {
  int lane = threadIdx.x & 31;
  const bf16* p = p0 + (long)(lane & 15) * ldb + ((lane >> 4) << 3);
  v8bf lo = *(const v8bf*)(p);
  v8bf hi = *(const v8bf*)(p + 16);
  return __builtin_shufflevector(lo, hi, 0,1,2,3,4,5,6,7,8,9,10,11,12,13,14,15);
}

// One wave computes a 32x32 f32 tile of A(MxK) @ Bt(NxK)^T, direct from L2.
__device__ __forceinline__ void wave_gemm32x32(const bf16* __restrict__ A, long lda,
                                               const bf16* __restrict__ Bt, long ldb,
                                               int K, v8f acc[2][2]) {
  #pragma unroll 2
  for (int k = 0; k < K; k += 32) {
    if (k + 32 < K) {                       // gfx1250 global_prefetch_b8 path
      __builtin_prefetch(A + k + 32, 0, 1);
      __builtin_prefetch(Bt + k + 32, 0, 1);
    }
    v16bf a0 = frag_a(A + k, lda);
    v16bf a1 = frag_a(A + 16 * lda + k, lda);
    v16bf b0 = frag_b(Bt + k, ldb);
    v16bf b1 = frag_b(Bt + 16 * ldb + k, ldb);
    acc[0][0] = __builtin_amdgcn_wmma_f32_16x16x32_bf16(false, a0, false, b0, (short)0, acc[0][0], false, false);
    acc[0][1] = __builtin_amdgcn_wmma_f32_16x16x32_bf16(false, a0, false, b1, (short)0, acc[0][1], false, false);
    acc[1][0] = __builtin_amdgcn_wmma_f32_16x16x32_bf16(false, a1, false, b0, (short)0, acc[1][0], false, false);
    acc[1][1] = __builtin_amdgcn_wmma_f32_16x16x32_bf16(false, a1, false, b1, (short)0, acc[1][1], false, false);
  }
}
// C/D layout: acc[i][j] element r -> (m = i*16 + ((lane>>4)<<3) + r, n = j*16 + (lane&15))

// ------- 8-wave (256t) LDS-staged block GEMM: 128(M) x 64(N) tile -----------
// A,Bt pre-offset to (m0,0)/(n0,0); row stride = K elements. Async double
// buffering: step k+32 is in flight (ASYNCcnt) while WMMA consumes step k.
// LDS rows padded to 40 elems (80B = 5x16B): async B128 writes stay aligned
// and ds_load_b128 fragment reads hit 16 distinct banks. Global addresses
// advance incrementally (+64B/step) to keep the loop rolled and VGPRs low.
static constexpr int LDA_S = 40;
__device__ __forceinline__ void block_gemm_128x64(const bf16* __restrict__ A,
                                                  const bf16* __restrict__ Bt,
                                                  int K, v8f acc[2][2],
                                                  int* wrp, int* wcp) {
  __shared__ __align__(16) bf16 sA[2][128 * LDA_S];
  __shared__ __align__(16) bf16 sB[2][64 * LDA_S];
  const int t = threadIdx.x;
  const int wave = t >> 5;
  const int wr = wave >> 1, wc = wave & 1;   // 4x2 wave grid
  *wrp = wr; *wcp = wc;

  unsigned bA[2] = { (unsigned)(size_t)&sA[0][0], (unsigned)(size_t)&sA[1][0] };
  unsigned bB[2] = { (unsigned)(size_t)&sB[0][0], (unsigned)(size_t)&sB[1][0] };

  // A tile: 128 rows x 64B -> 512 B128 chunks (2/thread); B: 64 rows -> 256.
  const int c0 = t, c1 = t + 256;
  const unsigned la0 = (unsigned)((c0 >> 2) * (LDA_S * 2) + (c0 & 3) * 16);
  const unsigned la1 = (unsigned)((c1 >> 2) * (LDA_S * 2) + (c1 & 3) * 16);
  const unsigned lb  = (unsigned)((t  >> 2) * (LDA_S * 2) + (t  & 3) * 16);
  const bf16* ga0 = A  + (long)(c0 >> 2) * K + (c0 & 3) * 8;
  const bf16* ga1 = A  + (long)(c1 >> 2) * K + (c1 & 3) * 8;
  const bf16* gb  = Bt + (long)(t  >> 2) * K + (t  & 3) * 8;

  auto issue = [&](int buf) {
    async_b128(bA[buf] + la0, ga0);
    async_b128(bA[buf] + la1, ga1);
    async_b128(bB[buf] + lb,  gb);
    ga0 += 32; ga1 += 32; gb += 32;          // next 64B K-slice
  };

  issue(0);
  wait_async0();
  __syncthreads();
  int buf = 0;
  #pragma unroll 1
  for (int k = 0; k < K; k += 32) {
    if (k + 32 < K) issue(buf ^ 1);          // overlap with compute
    const bf16* As = &sA[buf][0] + (wr * 32) * LDA_S;
    const bf16* Bs = &sB[buf][0] + (wc * 32) * LDA_S;
    v16bf a0 = frag_a(As, LDA_S);
    v16bf a1 = frag_a(As + 16 * LDA_S, LDA_S);
    v16bf b0 = frag_b(Bs, LDA_S);
    v16bf b1 = frag_b(Bs + 16 * LDA_S, LDA_S);
    acc[0][0] = __builtin_amdgcn_wmma_f32_16x16x32_bf16(false, a0, false, b0, (short)0, acc[0][0], false, false);
    acc[0][1] = __builtin_amdgcn_wmma_f32_16x16x32_bf16(false, a0, false, b1, (short)0, acc[0][1], false, false);
    acc[1][0] = __builtin_amdgcn_wmma_f32_16x16x32_bf16(false, a1, false, b0, (short)0, acc[1][0], false, false);
    acc[1][1] = __builtin_amdgcn_wmma_f32_16x16x32_bf16(false, a1, false, b1, (short)0, acc[1][1], false, false);
    wait_async0();        // next-step tiles have landed
    __syncthreads();      // everyone done reading buf before it is refilled
    buf ^= 1;
  }
}

// ---------------- elementwise convert f32 -> bf16 ---------------------------
__global__ void k_cvt(const float* __restrict__ in, bf16* __restrict__ out, long n) {
  long i = (long)blockIdx.x * blockDim.x + threadIdx.x;
  long st = (long)gridDim.x * blockDim.x;
  for (; i < n; i += st) out[i] = (bf16)in[i];
}

// ---------------- batched tiled transpose (+convert) to bf16 ---------------
// in[z][r][c] (RxC row-major per slab) -> out[z][c][r]
template <typename T>
__global__ void k_transpose_cvt(const T* __restrict__ in, bf16* __restrict__ out,
                                int R, int C) {
  __shared__ bf16 tile[32][33];
  long ib = (long)blockIdx.z * R * C;
  int r0 = blockIdx.y * 32, c0 = blockIdx.x * 32;
  int tx = threadIdx.x, ty = threadIdx.y;          // block (32, 8)
  #pragma unroll
  for (int i = 0; i < 32; i += 8)
    tile[ty + i][tx] = (bf16)in[ib + (long)(r0 + ty + i) * C + c0 + tx];
  __syncthreads();
  #pragma unroll
  for (int i = 0; i < 32; i += 8)
    out[ib + (long)(c0 + ty + i) * R + r0 + tx] = tile[tx][ty + i];
}

// ------- QKV projection (LDS-staged): P = (X @ W^T + b) * scale -------------
// X bf16 [4096 x 1024], W bf16 [1024 x 1024]; out bf16 [B,H,L,DH]
__global__ void __launch_bounds__(256)
k_proj(const bf16* __restrict__ X, const bf16* __restrict__ W,
       const float* __restrict__ bias, bf16* __restrict__ outp, float scale) {
  int lane = threadIdx.x & 31;
  int n0 = blockIdx.x * 64, m0 = blockIdx.y * 128;
  v8f acc[2][2] = {};
  int wr, wc;
  block_gemm_128x64(X + (long)m0 * Dn, W + (long)n0 * Dn, Dn, acc, &wr, &wc);
  #pragma unroll
  for (int i = 0; i < 2; ++i)
    #pragma unroll
    for (int j = 0; j < 2; ++j)
      #pragma unroll
      for (int r = 0; r < 8; ++r) {
        int m = m0 + wr * 32 + i * 16 + ((lane >> 4) << 3) + r;  // b*L + l
        int n = n0 + wc * 32 + j * 16 + (lane & 15);             // h*DH + d
        float v = (acc[i][j][r] + bias[n]) * scale;
        int b = m >> 10, l = m & 1023, h = n >> 6, d = n & 63;
        outp[(((long)(b * Hn + h)) * Ln + l) * DHn + d] = (bf16)v;
      }
}

// ---------------- content logits: S[bh] = q[bh] @ k[bh]^T -------------------
__global__ void k_qk(const bf16* __restrict__ qp, const bf16* __restrict__ kp,
                     float* __restrict__ logits) {
  int lane = threadIdx.x;
  int n0 = blockIdx.x * 32, m0 = blockIdx.y * 32, bh = blockIdx.z;
  const bf16* A  = qp + (long)bh * Ln * DHn + (long)m0 * DHn;
  const bf16* Bt = kp + (long)bh * Ln * DHn + (long)n0 * DHn;
  v8f acc[2][2] = {};
  wave_gemm32x32(A, DHn, Bt, DHn, DHn, acc);
  float* outb = logits + (long)bh * Ln * Ln;
  #pragma unroll
  for (int i = 0; i < 2; ++i)
    #pragma unroll
    for (int j = 0; j < 2; ++j)
      #pragma unroll
      for (int r = 0; r < 8; ++r) {
        int m = m0 + i * 16 + ((lane >> 4) << 3) + r;
        int n = n0 + j * 16 + (lane & 15);
        outb[(long)m * Ln + n] = acc[i][j][r];
      }
}

// ---- relative-key logits, per q: QR = Qall_q(64xDH) @ rel_k[q]^T(DHx1024) --
// rel_k is read exactly once (shared over all 64 bh). logits += QR.
__global__ void k_qr(const bf16* __restrict__ qp, const bf16* __restrict__ relk,
                     float* __restrict__ logits) {
  int lane = threadIdx.x;
  int n0 = blockIdx.x * 32, m0 = blockIdx.y * 32, q = blockIdx.z;
  const bf16* A  = qp + (long)m0 * Ln * DHn + (long)q * DHn;      // A[bh][d]
  const bf16* Bt = relk + ((long)q * Ln + n0) * DHn;              // Bt[kk][d]
  v8f acc[2][2] = {};
  wave_gemm32x32(A, (long)Ln * DHn, Bt, DHn, DHn, acc);
  #pragma unroll
  for (int i = 0; i < 2; ++i)
    #pragma unroll
    for (int j = 0; j < 2; ++j)
      #pragma unroll
      for (int r = 0; r < 8; ++r) {
        int bh = m0 + i * 16 + ((lane >> 4) << 3) + r;
        int kk = n0 + j * 16 + (lane & 15);
        float* p = logits + (long)bh * Ln * Ln + (long)q * Ln + kk;
        *p += acc[i][j][r];
      }
}

// ---------------- masked softmax, in place: f32 logits -> bf16 alpha --------
// alpha row (bh,q) is written as bf16 into the first half of its own f32 row,
// i.e. alpha has element stride 2*Ln = 2048 between rows.
__global__ void k_softmax(float* __restrict__ logits,
                          const unsigned char* __restrict__ mask) {
  const int q = blockIdx.x, bh = blockIdx.y, b = bh >> 4;
  float* row = logits + ((long)bh * Ln + q) * Ln;
  const unsigned char* mrow = mask + ((long)b * Ln + q) * Ln;
  bf16* arow = (bf16*)row;
  const int t = threadIdx.x;                      // 256 threads = 8 waves
  float v[4], lmax = -3.0e38f;
  #pragma unroll
  for (int i = 0; i < 4; ++i) {
    int k = t + i * 256;
    float x = row[k];
    x = mrow[k] ? -10000.0f : x;
    v[i] = x;
    lmax = fmaxf(lmax, x);
  }
  __shared__ float sred[8];
  #pragma unroll
  for (int off = 16; off > 0; off >>= 1) lmax = fmaxf(lmax, __shfl_down(lmax, off, 32));
  if ((t & 31) == 0) sred[t >> 5] = lmax;
  __syncthreads();
  float bmax = sred[0];
  #pragma unroll
  for (int w = 1; w < 8; ++w) bmax = fmaxf(bmax, sred[w]);
  float lsum = 0.f;
  #pragma unroll
  for (int i = 0; i < 4; ++i) { v[i] = __expf(v[i] - bmax); lsum += v[i]; }
  #pragma unroll
  for (int off = 16; off > 0; off >>= 1) lsum += __shfl_down(lsum, off, 32);
  __syncthreads();                                // protect sred reuse
  if ((t & 31) == 0) sred[t >> 5] = lsum;
  __syncthreads();
  float bsum = 0.f;
  #pragma unroll
  for (int w = 0; w < 8; ++w) bsum += sred[w];
  float inv = 1.0f / bsum;
  #pragma unroll
  for (int i = 0; i < 4; ++i) arow[t + i * 256] = (bf16)(v[i] * inv);
}

// ---------------- AV: per bh, alpha(1024x1024) @ v(1024xDH) -> ctxA ---------
// vT is [bh][DH][L] so B-operand is contiguous along K and L2-resident.
__global__ void k_av(const bf16* __restrict__ alpha, const bf16* __restrict__ vT,
                     float* __restrict__ ctxA) {
  int lane = threadIdx.x;
  int n0 = blockIdx.x * 32, m0 = blockIdx.y * 32, bh = blockIdx.z;
  const bf16* A  = alpha + ((long)bh * Ln + m0) * 2048;   // lda 2048 (f32 row alias)
  const bf16* Bt = vT + ((long)bh * DHn + n0) * Ln;
  v8f acc[2][2] = {};
  wave_gemm32x32(A, 2048, Bt, Ln, Ln, acc);
  int b = bh >> 4, h = bh & 15;
  #pragma unroll
  for (int i = 0; i < 2; ++i)
    #pragma unroll
    for (int j = 0; j < 2; ++j)
      #pragma unroll
      for (int r = 0; r < 8; ++r) {
        int m = m0 + i * 16 + ((lane >> 4) << 3) + r;     // query row l
        int n = n0 + j * 16 + (lane & 15);                // d
        ctxA[((long)b * Ln + m) * Dn + h * DHn + n] = acc[i][j][r];
      }
}

// ---- AR, per q: alpha[:,q,:](64x1024) @ rel_v[q](1024xDH) -> ctxB ----------
// relvT is [q][DH][L]; rel_v read exactly once.
__global__ void k_ar(const bf16* __restrict__ alpha, const bf16* __restrict__ relvT,
                     float* __restrict__ ctxB) {
  int lane = threadIdx.x;
  int n0 = blockIdx.x * 32, m0 = blockIdx.y * 32, q = blockIdx.z;
  const bf16* A  = alpha + (long)q * 2048 + (long)m0 * Ln * 2048;  // A[bh][k]
  const bf16* Bt = relvT + ((long)q * DHn + n0) * Ln;
  v8f acc[2][2] = {};
  wave_gemm32x32(A, (long)Ln * 2048, Bt, Ln, Ln, acc);
  #pragma unroll
  for (int i = 0; i < 2; ++i)
    #pragma unroll
    for (int j = 0; j < 2; ++j)
      #pragma unroll
      for (int r = 0; r < 8; ++r) {
        int bh = m0 + i * 16 + ((lane >> 4) << 3) + r;
        int n  = n0 + j * 16 + (lane & 15);               // d
        int b = bh >> 4, h = bh & 15;
        ctxB[((long)b * Ln + q) * Dn + h * DHn + n] = acc[i][j][r];
      }
}

// ---------------- ctx = bf16(ctxA + ctxB) -----------------------------------
__global__ void k_addcvt(const float* __restrict__ a, const float* __restrict__ b,
                         bf16* __restrict__ out, long n) {
  long i = (long)blockIdx.x * blockDim.x + threadIdx.x;
  long st = (long)gridDim.x * blockDim.x;
  for (; i < n; i += st) out[i] = (bf16)(a[i] + b[i]);
}

// ------- output projection (LDS-staged): out = ctx @ Wo^T + bo --------------
__global__ void __launch_bounds__(256)
k_outproj(const bf16* __restrict__ ctx, const bf16* __restrict__ Wo,
          const float* __restrict__ bo, float* __restrict__ out) {
  int lane = threadIdx.x & 31;
  int n0 = blockIdx.x * 64, m0 = blockIdx.y * 128;
  v8f acc[2][2] = {};
  int wr, wc;
  block_gemm_128x64(ctx + (long)m0 * Dn, Wo + (long)n0 * Dn, Dn, acc, &wr, &wc);
  #pragma unroll
  for (int i = 0; i < 2; ++i)
    #pragma unroll
    for (int j = 0; j < 2; ++j)
      #pragma unroll
      for (int r = 0; r < 8; ++r) {
        int m = m0 + wr * 32 + i * 16 + ((lane >> 4) << 3) + r;
        int n = n0 + wc * 32 + j * 16 + (lane & 15);
        out[(long)m * Dn + n] = acc[i][j][r] + bo[n];
      }
}

// ---------------------------------------------------------------------------
extern "C" void kernel_launch(void* const* d_in, const int* in_sizes, int n_in,
                              void* d_out, int out_size, void* d_ws, size_t ws_size,
                              hipStream_t stream) {
  const float* key   = (const float*)d_in[0];
  const float* value = (const float*)d_in[1];
  const float* query = (const float*)d_in[2];
  const unsigned char* mask = (const unsigned char*)d_in[3];
  const float* rel_k = (const float*)d_in[4];
  const float* rel_v = (const float*)d_in[5];
  const float* Wk = (const float*)d_in[6];
  const float* bk = (const float*)d_in[7];
  const float* Wv = (const float*)d_in[8];
  const float* bv = (const float*)d_in[9];
  const float* Wq = (const float*)d_in[10];
  const float* bq = (const float*)d_in[11];
  const float* Wo = (const float*)d_in[12];
  const float* bo = (const float*)d_in[13];
  float* out = (float*)d_out;

  const long XN = (long)Bn * Ln * Dn;          // 4,194,304
  const long WN = (long)Dn * Dn;               // 1,048,576
  const long RN = (long)Ln * Ln * DHn;         // 67,108,864
  const long PN = (long)BHn * Ln * DHn;        // 4,194,304
  const long LG = (long)BHn * Ln * Ln;         // 67,108,864

  char* ws = (char*)d_ws;
  auto alloc = [&](long elems, int esz) {
    char* p = ws;
    ws += ((long)elems * esz + 255) & ~255L;
    return p;
  };
  bf16* xk    = (bf16*)alloc(XN, 2);
  bf16* xv    = (bf16*)alloc(XN, 2);
  bf16* xq    = (bf16*)alloc(XN, 2);
  bf16* wkb   = (bf16*)alloc(WN, 2);
  bf16* wvb   = (bf16*)alloc(WN, 2);
  bf16* wqb   = (bf16*)alloc(WN, 2);
  bf16* wob   = (bf16*)alloc(WN, 2);
  bf16* relkb = (bf16*)alloc(RN, 2);
  bf16* relvT = (bf16*)alloc(RN, 2);
  bf16* kp    = (bf16*)alloc(PN, 2);
  bf16* vp    = (bf16*)alloc(PN, 2);
  bf16* qp    = (bf16*)alloc(PN, 2);
  bf16* vT    = (bf16*)alloc(PN, 2);
  float* logits = (float*)alloc(LG, 4);        // bf16 alpha aliases this in-place
  float* ctxA = (float*)alloc(XN, 4);
  float* ctxB = (float*)alloc(XN, 4);
  bf16* ctxbf = (bf16*)alloc(XN, 2);

  dim3 cvtg(4096), cvtb(256);
  hipLaunchKernelGGL(k_cvt, cvtg, cvtb, 0, stream, key,   xk,  XN);
  hipLaunchKernelGGL(k_cvt, cvtg, cvtb, 0, stream, value, xv,  XN);
  hipLaunchKernelGGL(k_cvt, cvtg, cvtb, 0, stream, query, xq,  XN);
  hipLaunchKernelGGL(k_cvt, dim3(1024), cvtb, 0, stream, Wk, wkb, WN);
  hipLaunchKernelGGL(k_cvt, dim3(1024), cvtb, 0, stream, Wv, wvb, WN);
  hipLaunchKernelGGL(k_cvt, dim3(1024), cvtb, 0, stream, Wq, wqb, WN);
  hipLaunchKernelGGL(k_cvt, dim3(1024), cvtb, 0, stream, Wo, wob, WN);
  hipLaunchKernelGGL(k_cvt, dim3(8192), cvtb, 0, stream, rel_k, relkb, RN);
  // rel_v [q][k][d] -> relvT [q][d][k]  (f32 in, bf16 out, LDS tile transpose)
  hipLaunchKernelGGL((k_transpose_cvt<float>), dim3(2, 32, 1024), dim3(32, 8), 0, stream,
                     rel_v, relvT, Ln, DHn);

  // projections (async-LDS staged): q scaled by 1/sqrt(DH) = 0.125
  dim3 pg(Dn / 64, (Bn * Ln) / 128), pb(256);
  hipLaunchKernelGGL(k_proj, pg, pb, 0, stream, xk, wkb, bk, kp, 1.0f);
  hipLaunchKernelGGL(k_proj, pg, pb, 0, stream, xv, wvb, bv, vp, 1.0f);
  hipLaunchKernelGGL(k_proj, pg, pb, 0, stream, xq, wqb, bq, qp, 0.125f);
  // vp [bh][l][d] -> vT [bh][d][l]
  hipLaunchKernelGGL((k_transpose_cvt<bf16>), dim3(2, 32, BHn), dim3(32, 8), 0, stream,
                     vp, vT, Ln, DHn);

  dim3 wv1(32);
  hipLaunchKernelGGL(k_qk, dim3(Ln / 32, Ln / 32, BHn), wv1, 0, stream, qp, kp, logits);
  hipLaunchKernelGGL(k_qr, dim3(Ln / 32, BHn / 32, Ln), wv1, 0, stream, qp, relkb, logits);
  hipLaunchKernelGGL(k_softmax, dim3(Ln, BHn), dim3(256), 0, stream, logits, mask);

  const bf16* alpha = (const bf16*)logits;     // row stride 2048 bf16 elems
  hipLaunchKernelGGL(k_av, dim3(DHn / 32, Ln / 32, BHn), wv1, 0, stream, alpha, vT, ctxA);
  hipLaunchKernelGGL(k_ar, dim3(DHn / 32, BHn / 32, Ln), wv1, 0, stream, alpha, relvT, ctxB);
  hipLaunchKernelGGL(k_addcvt, cvtg, cvtb, 0, stream, ctxA, ctxB, ctxbf, XN);
  hipLaunchKernelGGL(k_outproj, pg, pb, 0, stream, ctxbf, wob, bo, out);
}
